// Attention_34230889349528
// MI455X (gfx1250) — matrix operand order repaired and checked by reading, not compiled, not run on hardware
//
#include <hip/hip_runtime.h>
#include <math.h>

#define DIMC 768
#define NHEAD 12
#define HDIM 64
#define SEQ 197
#define NBATCH 64
#define TT 209      // SEQ + NHEAD
#define TP 224      // 14 * 16 padded keys
#define QT 14       // query tiles of 16
#define NT 14       // key tiles of 16
#define C3 2304

typedef __bf16 bf16_t;
typedef __attribute__((ext_vector_type(16))) __bf16 v16bf;
typedef __attribute__((ext_vector_type(8)))  float  v8f;

__device__ __forceinline__ bf16_t f2bf(float f) {
  unsigned u = __builtin_bit_cast(unsigned, f);
  u += 0x7FFFu + ((u >> 16) & 1u);
  unsigned short h = (unsigned short)(u >> 16);
  return __builtin_bit_cast(bf16_t, h);
}

// A-fragment (16x32 bf16, M x K): lanes 0-15 = rows, element e holds
// K = e + (e<8 ? (half?8:0) : (half?16:8))  => two contiguous 8-runs.
__device__ __forceinline__ void load_afrag(v16bf& a, const bf16_t* __restrict__ row,
                                           int k0, int half) {
  int s0 = k0 + (half ? 8 : 0);
  int s1 = k0 + (half ? 24 : 16);
#pragma unroll
  for (int e = 0; e < 8; ++e) a[e] = row[s0 + e];
#pragma unroll
  for (int e = 0; e < 8; ++e) a[8 + e] = row[s1 + e];
}

// B-fragment (32x16 bf16, K x N): lane r = column, same K order as A-fragment
// (element e: K = k0 + e + (e<8 ? (half?8:0) : (half?16:8))).
__device__ __forceinline__ void load_bfrag(v16bf& b, const bf16_t* __restrict__ col,
                                           int k0, int half) {
#pragma unroll
  for (int e = 0; e < 16; ++e) b[e] = col[k0 + (e < 8 ? 8 * half + e : 16 + 8 * half + (e - 8))];
}

// ---- generic WMMA GEMM, 1x4 N-blocked: C[MxN] = A[MxK] @ Bt[NxK]^T + bias --
__global__ __launch_bounds__(256)
void wmma_gemm_kernel(const bf16_t* __restrict__ A, const bf16_t* __restrict__ Bt,
                      const float* __restrict__ bias,
                      bf16_t* __restrict__ Cbf, float* __restrict__ Cf,
                      int M, int N, int K) {
  int lane = threadIdx.x & 31;
  int wave = threadIdx.x >> 5;
  int half = lane >> 4;
  int r    = lane & 15;
  int tilesN4 = N >> 6;                       // N multiple of 64
  long tilesM = (M + 15) >> 4;
  long tile = (long)blockIdx.x * (blockDim.x >> 5) + wave;
  if (tile >= tilesM * tilesN4) return;
  int tm = (int)(tile / tilesN4);
  int n0 = (int)(tile % tilesN4) * 64;

  int rowA = tm * 16 + r;
  int rowAc = rowA < M ? rowA : (M - 1);
  const bf16_t* Arow = A + (size_t)rowAc * K;
  const bf16_t* B0 = Bt + (size_t)(n0 + r) * K;

  v8f acc0 = {}, acc1 = {}, acc2 = {}, acc3 = {};
  for (int k0 = 0; k0 < K; k0 += 32) {
    __builtin_prefetch(Arow + k0 + 256, 0, 1);           // global_prefetch_b8
    v16bf a, b0, b1, b2, b3;
    load_afrag(a, Arow, k0, half);
    load_bfrag(b0, B0,                      k0, half);
    load_bfrag(b1, B0 + (size_t)16 * K,     k0, half);
    load_bfrag(b2, B0 + (size_t)32 * K,     k0, half);
    load_bfrag(b3, B0 + (size_t)48 * K,     k0, half);
    acc0 = __builtin_amdgcn_wmma_f32_16x16x32_bf16(false, a, false, b0, (short)0, acc0, false, false);
    acc1 = __builtin_amdgcn_wmma_f32_16x16x32_bf16(false, a, false, b1, (short)0, acc1, false, false);
    acc2 = __builtin_amdgcn_wmma_f32_16x16x32_bf16(false, a, false, b2, (short)0, acc2, false, false);
    acc3 = __builtin_amdgcn_wmma_f32_16x16x32_bf16(false, a, false, b3, (short)0, acc3, false, false);
  }
  v8f accs[4] = {acc0, acc1, acc2, acc3};
#pragma unroll
  for (int j = 0; j < 4; ++j) {
    float bn = bias[n0 + j * 16 + r];
#pragma unroll
    for (int i = 0; i < 8; ++i) {
      int m = tm * 16 + i + 8 * half;       // C/D: VGPR i -> M=i(+8), N=lane%16
      if (m < M) {
        float v = accs[j][i] + bn;
        size_t idx = (size_t)m * N + n0 + j * 16 + r;
        if (Cf) Cf[idx] = v;
        else    Cbf[idx] = f2bf(v);
      }
    }
  }
}

// ---------------- attention: one block per (b,h), 4 waves -------------------
__global__ __launch_bounds__(128)
void attn_kernel(const bf16_t* __restrict__ qkv, bf16_t* __restrict__ outbf) {
  __shared__ bf16_t Vt[HDIM * TP];          // 28 KB: V transposed (d-major)
  __shared__ bf16_t Pbuf[4][16 * TP];       // 28 KB: per-wave softmax probs
  int b = blockIdx.x / NHEAD;
  int h = blockIdx.x % NHEAD;
  int lane = threadIdx.x & 31;
  int wave = threadIdx.x >> 5;
  int half = lane >> 4;
  int r    = lane & 15;
  const float scale = 0.125f;               // HD^-0.5

  // cooperative transposed fill of V (keys >= TT zero-padded)
  for (int idx = threadIdx.x; idx < TP * HDIM; idx += 128) {
    int t = idx >> 6, d = idx & 63;         // read-coalesced over d
    float v = 0.f;
    if (t < TT) {
      bf16_t raw = qkv[((size_t)(b * TT + t) * 3 + 2) * DIMC + h * HDIM + d];
      Vt[d * TP + t] = raw;
    } else {
      Vt[d * TP + t] = f2bf(v);
    }
  }
  __syncthreads();

  bf16_t* P = Pbuf[wave];
  for (int qt = wave; qt < QT; qt += 4) {
    // Q A-fragments are identical for every key tile: load once.
    int qrow = qt * 16 + r;  if (qrow >= TT) qrow = TT - 1;
    const bf16_t* Arow = qkv + ((size_t)(b * TT + qrow) * 3 + 0) * DIMC + h * HDIM;
    v16bf aq0, aq1;
    load_afrag(aq0, Arow, 0, half);
    load_afrag(aq1, Arow, 32, half);

    // ---- pass A: row maxima ----
    float rowmax[8];
#pragma unroll
    for (int i = 0; i < 8; ++i) rowmax[i] = -3.0e38f;
    for (int nt = 0; nt < NT; ++nt) {
      int key = nt * 16 + r;  int keyc = key < TT ? key : TT - 1;
      const bf16_t* Brow = qkv + ((size_t)(b * TT + keyc) * 3 + 1) * DIMC + h * HDIM;
      v16bf bk0, bk1;
      load_bfrag(bk0, Brow, 0, half);
      load_bfrag(bk1, Brow, 32, half);
      v8f acc = {};
      acc = __builtin_amdgcn_wmma_f32_16x16x32_bf16(false, aq0, false, bk0, (short)0, acc, false, false);
      acc = __builtin_amdgcn_wmma_f32_16x16x32_bf16(false, aq1, false, bk1, (short)0, acc, false, false);
      bool valid = key < TT;
#pragma unroll
      for (int i = 0; i < 8; ++i) {
        float s = valid ? acc[i] * scale : -3.0e38f;
#pragma unroll
        for (int m = 8; m >= 1; m >>= 1) s = fmaxf(s, __shfl_xor(s, m, 16));
        rowmax[i] = fmaxf(rowmax[i], s);
      }
    }
    // ---- pass B: exp, row sums, stage P (bf16) in LDS ----
    float rowsum[8];
#pragma unroll
    for (int i = 0; i < 8; ++i) rowsum[i] = 0.0f;
    for (int nt = 0; nt < NT; ++nt) {
      int key = nt * 16 + r;  int keyc = key < TT ? key : TT - 1;
      const bf16_t* Brow = qkv + ((size_t)(b * TT + keyc) * 3 + 1) * DIMC + h * HDIM;
      v16bf bk0, bk1;
      load_bfrag(bk0, Brow, 0, half);
      load_bfrag(bk1, Brow, 32, half);
      v8f acc = {};
      acc = __builtin_amdgcn_wmma_f32_16x16x32_bf16(false, aq0, false, bk0, (short)0, acc, false, false);
      acc = __builtin_amdgcn_wmma_f32_16x16x32_bf16(false, aq1, false, bk1, (short)0, acc, false, false);
      bool valid = key < TT;
#pragma unroll
      for (int i = 0; i < 8; ++i) {
        float p = valid ? __expf(acc[i] * scale - rowmax[i]) : 0.0f;
        P[(i + 8 * half) * TP + nt * 16 + r] = f2bf(p);
        float ps = p;
#pragma unroll
        for (int m = 8; m >= 1; m >>= 1) ps += __shfl_xor(ps, m, 16);
        rowsum[i] += ps;
      }
    }
    asm volatile("s_wait_dscnt 0" ::: "memory");  // wave's LDS stores visible
    // ---- PV: O(16x64) = P(16x224) @ V(224x64), normalize by rowsum ----
    for (int dt = 0; dt < 4; ++dt) {
      v8f acc = {};
#pragma unroll 1
      for (int ks = 0; ks < 7; ++ks) {
        int k0 = ks * 32;
        v16bf a, bfrag;
        load_afrag(a, P + r * TP, k0, half);
        load_bfrag(bfrag, Vt + (dt * 16 + r) * TP, k0, half);
        acc = __builtin_amdgcn_wmma_f32_16x16x32_bf16(false, a, false, bfrag,
                                                      (short)0, acc, false, false);
      }
#pragma unroll
      for (int i = 0; i < 8; ++i) {
        int m = qt * 16 + i + 8 * half;
        if (m < TT) {
          float inv = 1.0f / fmaxf(rowsum[i], 1e-30f);
          outbf[((size_t)(b * TT + m)) * DIMC + h * HDIM + dt * 16 + r] =
              f2bf(acc[i] * inv);
        }
      }
    }
  }
}

// ---------------- small prep / epilogue kernels ----------------------------
__global__ void convw_kernel(const float* __restrict__ w, bf16_t* __restrict__ wt,
                             int K, int N) {               // w KxN -> wt NxK bf16
  size_t i = (size_t)blockIdx.x * blockDim.x + threadIdx.x;
  if (i >= (size_t)K * N) return;
  int n = (int)(i / K), k = (int)(i % K);
  wt[i] = f2bf(w[(size_t)k * N + n]);
}

__global__ void xcopy_kernel(const float* __restrict__ x, bf16_t* __restrict__ xc) {
  size_t i = (size_t)blockIdx.x * blockDim.x + threadIdx.x;
  if (i >= (size_t)NBATCH * SEQ * DIMC) return;
  size_t b = i / ((size_t)SEQ * DIMC), rem = i % ((size_t)SEQ * DIMC);
  xc[b * TT * DIMC + rem] = f2bf(x[i]);
}

__global__ void colmean_kernel(const float* __restrict__ x, float* __restrict__ cm) {
  int i = blockIdx.x * blockDim.x + threadIdx.x;
  if (i >= NBATCH * DIMC) return;
  int b = i / DIMC, c = i % DIMC;
  float s = 0.f;
  for (int n = 0; n < SEQ; ++n) s += x[((size_t)b * SEQ + n) * DIMC + c];
  cm[i] = s * (1.0f / SEQ);
}

__global__ void ht_kernel(const float* __restrict__ cm, const float* __restrict__ htw,
                          const float* __restrict__ htb, float* __restrict__ htraw) {
  int i = blockIdx.x * blockDim.x + threadIdx.x;
  if (i >= NBATCH * NHEAD * DIMC) return;
  int c = i % DIMC, bh = i / DIMC;
  int h1 = bh % NHEAD, b = bh / NHEAD;
  float s = htb[c];
  for (int d = 0; d < HDIM; ++d)
    s += cm[b * DIMC + h1 * HDIM + d] * htw[(size_t)d * DIMC + c];
  htraw[i] = s;
}

__global__ void lngelu_kernel(const float* __restrict__ htraw,
                              const float* __restrict__ lng, const float* __restrict__ lnb,
                              const float* __restrict__ pos, bf16_t* __restrict__ xc) {
  int i = blockIdx.x * blockDim.x + threadIdx.x;
  if (i >= NBATCH * NHEAD * NHEAD) return;
  int h2 = i % NHEAD, t = i / NHEAD;
  int h1 = t % NHEAD, b = t / NHEAD;
  const float* p = htraw + ((size_t)(b * NHEAD + h1)) * DIMC + h2 * HDIM;
  float mu = 0.f;
  for (int d = 0; d < HDIM; ++d) mu += p[d];
  mu *= (1.0f / HDIM);
  float var = 0.f;
  for (int d = 0; d < HDIM; ++d) { float dv = p[d] - mu; var += dv * dv; }
  var *= (1.0f / HDIM);
  float rs = rsqrtf(var + 1e-5f);
  for (int d = 0; d < HDIM; ++d) {
    float y = (p[d] - mu) * rs * lng[d] + lnb[d];
    float g = 0.5f * y * (1.0f + erff(y * 0.70710678118654752f));
    float o = g + pos[(size_t)h1 * DIMC + h2 * HDIM + d];
    xc[((size_t)(b * TT + SEQ + h1)) * DIMC + h2 * HDIM + d] = f2bf(o);
  }
}

__global__ void final_kernel(const float* __restrict__ po, float* __restrict__ out) {
  size_t i = (size_t)blockIdx.x * blockDim.x + threadIdx.x;
  if (i >= (size_t)NBATCH * SEQ * DIMC) return;
  int c = (int)(i % DIMC);
  size_t t = i / DIMC;
  int n = (int)(t % SEQ), b = (int)(t / SEQ);
  size_t base = (size_t)b * TT * DIMC;
  float v = po[base + (size_t)n * DIMC + c];
  if (n == 0) {
    float s = 0.f;
    for (int j = 0; j < NHEAD; ++j) s += po[base + (size_t)(SEQ + j) * DIMC + c];
    v += s * (1.0f / NHEAD);
  }
  out[i] = v;
}

// ---------------------------------------------------------------------------
extern "C" void kernel_launch(void* const* d_in, const int* in_sizes, int n_in,
                              void* d_out, int out_size, void* d_ws, size_t ws_size,
                              hipStream_t stream) {
  const float* x      = (const float*)d_in[0];
  const float* qkv_w  = (const float*)d_in[1];
  const float* qkv_b  = (const float*)d_in[2];
  const float* proj_w = (const float*)d_in[3];
  const float* proj_b = (const float*)d_in[4];
  const float* ht_w   = (const float*)d_in[5];
  const float* ht_b   = (const float*)d_in[6];
  const float* ln_g   = (const float*)d_in[7];
  const float* ln_b   = (const float*)d_in[8];
  const float* pose   = (const float*)d_in[9];
  float* out = (float*)d_out;

  const size_t MROWS = (size_t)NBATCH * TT;   // 13376 (multiple of 16)
  size_t off = 0;
  auto take = [&](size_t bytes) {
    void* p = (char*)d_ws + off;
    off += (bytes + 255) & ~(size_t)255;
    return p;
  };
  bf16_t* qkvwT   = (bf16_t*)take((size_t)C3 * DIMC * 2);
  bf16_t* projwT  = (bf16_t*)take((size_t)DIMC * DIMC * 2);
  bf16_t* xc_bf   = (bf16_t*)take(MROWS * DIMC * 2);
  bf16_t* qkv_bf  = (bf16_t*)take(MROWS * C3 * 2);
  bf16_t* ao_bf   = (bf16_t*)take(MROWS * DIMC * 2);
  float*  projout = (float*) take(MROWS * DIMC * 4);
  float*  cmean   = (float*) take((size_t)NBATCH * DIMC * 4);
  float*  htraw   = (float*) take((size_t)NBATCH * NHEAD * DIMC * 4);

  const int TPB = 256;
  auto blocks = [](size_t n, int t) { return (unsigned)((n + t - 1) / t); };

  convw_kernel<<<blocks((size_t)DIMC * C3, TPB), TPB, 0, stream>>>(qkv_w, qkvwT, DIMC, C3);
  convw_kernel<<<blocks((size_t)DIMC * DIMC, TPB), TPB, 0, stream>>>(proj_w, projwT, DIMC, DIMC);
  xcopy_kernel<<<blocks((size_t)NBATCH * SEQ * DIMC, TPB), TPB, 0, stream>>>(x, xc_bf);
  colmean_kernel<<<blocks((size_t)NBATCH * DIMC, TPB), TPB, 0, stream>>>(x, cmean);
  ht_kernel<<<blocks((size_t)NBATCH * NHEAD * DIMC, TPB), TPB, 0, stream>>>(cmean, ht_w, ht_b, htraw);
  lngelu_kernel<<<blocks((size_t)NBATCH * NHEAD * NHEAD, TPB), TPB, 0, stream>>>(htraw, ln_g, ln_b, pose, xc_bf);

  { // qkv GEMM: (MROWS x 768) @ (768 x 2304), 16x64 tiles per wave
    long tiles = (long)(MROWS / 16) * (C3 / 64);
    wmma_gemm_kernel<<<blocks((size_t)tiles, 8), TPB, 0, stream>>>(
        xc_bf, qkvwT, qkv_b, qkv_bf, (float*)nullptr, (int)MROWS, C3, DIMC);
  }
  attn_kernel<<<NBATCH * NHEAD, 128, 0, stream>>>(qkv_bf, ao_bf);
  { // proj GEMM: (MROWS x 768) @ (768 x 768) -> fp32
    long tiles = (long)(MROWS / 16) * (DIMC / 64);
    wmma_gemm_kernel<<<blocks((size_t)tiles, 8), TPB, 0, stream>>>(
        ao_bf, projwT, proj_b, (bf16_t*)nullptr, projout, (int)MROWS, DIMC, DIMC);
  }
  final_kernel<<<blocks((size_t)NBATCH * SEQ * DIMC, TPB), TPB, 0, stream>>>(projout, out);
}
